// Transformer_2920577761718
// MI455X (gfx1250) — compile-verified, matrix operand order checked by reading, not run on hardware
//
#include <hip/hip_runtime.h>
#include <hip/hip_bf16.h>

#define BDIM    4
#define TSEQ    1024
#define MAXT    1025
#define MP      1056          // padded seq (multiple of 32)
#define HDIM    1024
#define NHEADS  16
#define DHEAD   64
#define NLAYER  8
#define FFDIM   4096
#define SDIM    64
#define ADIM    32
#define OUTD    32
#define MROWS   (BDIM * MP)   // 4224
#define LNEPS   1e-6f

typedef __bf16 bf16x16 __attribute__((ext_vector_type(16)));
typedef float  f32x8   __attribute__((ext_vector_type(8)));
typedef unsigned int v4u __attribute__((ext_vector_type(4)));
typedef int v8i __attribute__((ext_vector_type(8)));
typedef int v4i __attribute__((ext_vector_type(4)));

union Frag { bf16x16 v; unsigned int u[8]; };
union Acc  { f32x8 v; float f[8]; };

static __device__ __forceinline__ unsigned short f2bf(float x) {
    unsigned int u = __float_as_uint(x);
    u += 0x7FFFu + ((u >> 16) & 1u);          // round-to-nearest-even
    return (unsigned short)(u >> 16);
}

// ---------------------------------------------------------------------------
// TDM: async 2-D bf16 tile load (tileK contiguous elems per row, tileN rows)
// from a [N][K] bf16 matrix into LDS at lds_off. D# per CDNA5 ISA.
// ---------------------------------------------------------------------------
static __device__ __forceinline__ void tdm_load_bf16_tile(
    unsigned lds_off, const unsigned short* gptr,
    unsigned tileK, unsigned tileN, unsigned Kfull, unsigned Nfull)
{
    const unsigned long long ga = (unsigned long long)(uintptr_t)gptr;
    v4u g0;
    g0.x = 0x1u;                                   // count=1 (valid), user mode
    g0.y = lds_off;                                // LDS byte address
    g0.z = (unsigned)(ga & 0xFFFFFFFFu);           // global_addr[31:0]
    g0.w = (unsigned)((ga >> 32) & 0x01FFFFFFu) | (2u << 30);  // addr[56:32]|type=2
    v8i g1;
    g1[0] = (int)(1u << 16);                       // data_size=1 (2 bytes)
    g1[1] = (int)((Kfull & 0xFFFFu) << 16);        // tensor_dim0[15:0]
    g1[2] = (int)((Kfull >> 16) | ((Nfull & 0xFFFFu) << 16));  // dim0 hi | dim1 lo
    g1[3] = (int)((Nfull >> 16) | (tileK << 16));  // dim1 hi | tile_dim0
    g1[4] = (int)tileN;                            // tile_dim1 (tile_dim2=0)
    g1[5] = (int)Kfull;                            // tensor_dim0_stride[31:0]
    g1[6] = 0;                                     // stride hi | dim1_stride lo
    g1[7] = 0;
    v4i gz = {0, 0, 0, 0};
#if defined(__clang_major__) && (__clang_major__ >= 23)
    v8i gz8 = {0, 0, 0, 0, 0, 0, 0, 0};
    __builtin_amdgcn_tensor_load_to_lds(g0, g1, gz, gz, gz8, 0);
#else
    __builtin_amdgcn_tensor_load_to_lds(g0, g1, gz, gz, 0);
#endif
}

// ---------------------------------------------------------------------------
// One-time weight prep: W[K][N] f32 -> Wt[N][K] bf16 (transpose via LDS tile)
// ---------------------------------------------------------------------------
__global__ __launch_bounds__(256)
void wconv_kernel(const float* __restrict__ src, unsigned short* __restrict__ dst,
                  int K, int N)
{
    __shared__ unsigned short tile[64][65];
    const int tid = threadIdx.x;
    const int nt = blockIdx.x * 64, kt = blockIdx.y * 64;
    const int r = tid >> 2, c0 = (tid & 3) * 16;
    #pragma unroll
    for (int j = 0; j < 16; j++)
        tile[r][c0 + j] = f2bf(src[(size_t)(kt + r) * N + nt + c0 + j]);
    __syncthreads();
    #pragma unroll
    for (int j = 0; j < 16; j++)
        dst[(size_t)(nt + r) * K + kt + c0 + j] = tile[c0 + j][r];
}

// ---------------------------------------------------------------------------
// Shared GEMM epilogue (wave tile 32x64 = 2x4 accum tiles)
// ---------------------------------------------------------------------------
template<bool OUT_BF16, bool DO_GELU, bool RESID>
static __device__ __forceinline__ void gemm_epilogue(
    Acc (&acc)[2][4], const float* bias, const float* resid, void* Cptr,
    int M, int N, int bm, int bn, int wm, int wn, int lane)
{
    const int l15 = lane & 15;
    #pragma unroll
    for (int i = 0; i < 2; i++) {
        #pragma unroll
        for (int j = 0; j < 4; j++) {
            const int col = bn + wn + j * 16 + l15;
            #pragma unroll
            for (int r = 0; r < 8; r++) {
                const int row = bm + wm + i * 16 + r + ((lane >> 4) << 3);
                float v = acc[i][j].f[r] + bias[col];
                if (DO_GELU) {
                    const float x = v;
                    v = 0.5f * x * (1.f + tanhf(0.7978845608028654f *
                                                (x + 0.044715f * x * x * x)));
                }
                if (RESID) v += resid[(size_t)row * N + col];
                if (row < M) {
                    if (OUT_BF16)
                        ((unsigned short*)Cptr)[(size_t)row * N + col] = f2bf(v);
                    else
                        ((float*)Cptr)[(size_t)row * N + col] = v;
                }
            }
        }
    }
}

// ---------------------------------------------------------------------------
// TDM-fed bf16 WMMA GEMM: C = act(A[M,K] @ WtT) ; Wt is [N][K] bf16.
// Block tile 64x256; wave tile 32x64; B tiles double-buffered via TDM.
// ---------------------------------------------------------------------------
template<bool A_BF16, bool OUT_BF16, bool DO_GELU, bool RESID>
__global__ __launch_bounds__(256)
void gemm_wmma_tdm(const void* __restrict__ Aptr, const unsigned short* __restrict__ Wt,
                   const float* __restrict__ bias, const float* __restrict__ resid,
                   void* __restrict__ Cptr, int M, int N, int K)
{
    __shared__ __align__(16) unsigned short As[64][32];       // [m][k]
    __shared__ __align__(16) unsigned short Bb[2][256][32];   // [n][k] x2 (TDM dest)

    const int tid  = threadIdx.x;
    const int lane = tid & 31;
    const int wave = tid >> 5;
    const int bm = blockIdx.y * 64;
    const int bn = blockIdx.x * 256;
    const int wm = (wave >> 2) * 32;
    const int wn = (wave & 3) * 64;
    const int l15 = lane & 15;
    const int kb  = (lane >> 4) * 8;

    Acc acc[2][4];
    for (int i = 0; i < 2; i++)
        for (int j = 0; j < 4; j++)
            for (int r = 0; r < 8; r++) acc[i][j].f[r] = 0.f;

    const int aRow = tid >> 2;          // 0..63
    const int aCol = (tid & 3) * 8;     // 0,8,16,24
    const unsigned short* wbase = Wt + (size_t)bn * K;

    if (wave == 0)                       // preload B tile 0
        tdm_load_bf16_tile((unsigned)(uintptr_t)&Bb[0][0][0], wbase,
                           32u, 256u, (unsigned)K, (unsigned)N);

    for (int k0 = 0; k0 < K; k0 += 32) {
        const int ibuf = (k0 >> 5) & 1;
        // --- A tile -> LDS (bf16) ---
        {
            const size_t gro = (size_t)(bm + aRow) * K + k0 + aCol;
            if (A_BF16) {
                const unsigned short* A16 = (const unsigned short*)Aptr;
                #pragma unroll
                for (int j = 0; j < 8; j++) As[aRow][aCol + j] = A16[gro + j];
            } else {
                const float* A32 = (const float*)Aptr;
                #pragma unroll
                for (int j = 0; j < 8; j++) As[aRow][aCol + j] = f2bf(A32[gro + j]);
            }
        }
        // --- wave 0: issue next B tile, wait for current ---
        if (wave == 0) {
            if (k0 + 32 < K) {
                tdm_load_bf16_tile((unsigned)(uintptr_t)&Bb[1 - ibuf][0][0],
                                   wbase + (k0 + 32), 32u, 256u,
                                   (unsigned)K, (unsigned)N);
                __builtin_amdgcn_s_wait_tensorcnt(1);
            } else {
                __builtin_amdgcn_s_wait_tensorcnt(0);
            }
        }
        __syncthreads();                 // A staged + current B tile resident

        Frag af[2], bf[4];
        #pragma unroll
        for (int i = 0; i < 2; i++) {
            const int r = wm + i * 16 + l15;
            #pragma unroll
            for (int j = 0; j < 4; j++) {
                af[i].u[j]     = *(const unsigned int*)&As[r][kb + 2 * j];
                af[i].u[4 + j] = *(const unsigned int*)&As[r][kb + 16 + 2 * j];
            }
        }
        #pragma unroll
        for (int i = 0; i < 4; i++) {
            const int c = wn + i * 16 + l15;
            #pragma unroll
            for (int j = 0; j < 4; j++) {
                bf[i].u[j]     = *(const unsigned int*)&Bb[ibuf][c][kb + 2 * j];
                bf[i].u[4 + j] = *(const unsigned int*)&Bb[ibuf][c][kb + 16 + 2 * j];
            }
        }
        #pragma unroll
        for (int i = 0; i < 2; i++)
            #pragma unroll
            for (int j = 0; j < 4; j++)
                acc[i][j].v = __builtin_amdgcn_wmma_f32_16x16x32_bf16(
                    false, af[i].v, false, bf[j].v, (short)0, acc[i][j].v,
                    false, false);
        __syncthreads();                 // done reading before next overwrite
    }
    gemm_epilogue<OUT_BF16, DO_GELU, RESID>(acc, bias, resid, Cptr,
                                            M, N, bm, bn, wm, wn, lane);
}

// ---------------------------------------------------------------------------
// Fallback GEMM (fp32 weights converted in-kernel) — used if ws too small.
// ---------------------------------------------------------------------------
template<bool A_BF16, bool OUT_BF16, bool DO_GELU, bool RESID>
__global__ __launch_bounds__(256)
void gemm_wmma(const void* __restrict__ Aptr, const float* __restrict__ W,
               const float* __restrict__ bias, const float* __restrict__ resid,
               void* __restrict__ Cptr, int M, int N, int K)
{
    __shared__ unsigned short As[64][32];
    __shared__ unsigned short Bs[256][32];

    const int tid  = threadIdx.x;
    const int lane = tid & 31;
    const int wave = tid >> 5;
    const int bm = blockIdx.y * 64;
    const int bn = blockIdx.x * 256;
    const int wm = (wave >> 2) * 32;
    const int wn = (wave & 3) * 64;
    const int l15 = lane & 15;
    const int kb  = (lane >> 4) * 8;

    Acc acc[2][4];
    for (int i = 0; i < 2; i++)
        for (int j = 0; j < 4; j++)
            for (int r = 0; r < 8; r++) acc[i][j].f[r] = 0.f;

    const int aRow = tid >> 2, aCol = (tid & 3) * 8;
    const int bKr  = tid >> 3, bNg  = (tid & 7) * 32;

    for (int k0 = 0; k0 < K; k0 += 32) {
        {
            const size_t gro = (size_t)(bm + aRow) * K + k0 + aCol;
            if (A_BF16) {
                const unsigned short* A16 = (const unsigned short*)Aptr;
                #pragma unroll
                for (int j = 0; j < 8; j++) As[aRow][aCol + j] = A16[gro + j];
            } else {
                const float* A32 = (const float*)Aptr;
                #pragma unroll
                for (int j = 0; j < 8; j++) As[aRow][aCol + j] = f2bf(A32[gro + j]);
            }
        }
        {
            const size_t gro = (size_t)(k0 + bKr) * N + bn + bNg;
            #pragma unroll
            for (int j = 0; j < 32; j++) Bs[bNg + j][bKr] = f2bf(W[gro + j]);
        }
        __syncthreads();

        Frag af[2], bf[4];
        #pragma unroll
        for (int i = 0; i < 2; i++) {
            const int r = wm + i * 16 + l15;
            #pragma unroll
            for (int j = 0; j < 4; j++) {
                af[i].u[j]     = *(const unsigned int*)&As[r][kb + 2 * j];
                af[i].u[4 + j] = *(const unsigned int*)&As[r][kb + 16 + 2 * j];
            }
        }
        #pragma unroll
        for (int i = 0; i < 4; i++) {
            const int c = wn + i * 16 + l15;
            #pragma unroll
            for (int j = 0; j < 4; j++) {
                bf[i].u[j]     = *(const unsigned int*)&Bs[c][kb + 2 * j];
                bf[i].u[4 + j] = *(const unsigned int*)&Bs[c][kb + 16 + 2 * j];
            }
        }
        #pragma unroll
        for (int i = 0; i < 2; i++)
            #pragma unroll
            for (int j = 0; j < 4; j++)
                acc[i][j].v = __builtin_amdgcn_wmma_f32_16x16x32_bf16(
                    false, af[i].v, false, bf[j].v, (short)0, acc[i][j].v,
                    false, false);
        __syncthreads();
    }
    gemm_epilogue<OUT_BF16, DO_GELU, RESID>(acc, bias, resid, Cptr,
                                            M, N, bm, bn, wm, wn, lane);
}

// ---------------------------------------------------------------------------
// Causal flash attention: one wave per 16-row query tile, 32-key blocks.
// ---------------------------------------------------------------------------
__global__ __launch_bounds__(256)
void attn_wmma(const unsigned short* __restrict__ Q,
               const unsigned short* __restrict__ Kb,
               const unsigned short* __restrict__ V,
               float* __restrict__ O)
{
    __shared__ unsigned short Pb[8][16][32];
    __shared__ unsigned short Vt[8][64][32];

    const int tid = threadIdx.x, lane = tid & 31, w = tid >> 5;
    const int bh = blockIdx.x;
    const int b = bh / NHEADS, hh = bh % NHEADS;
    const int m0 = blockIdx.y * 128 + w * 16;
    if (m0 >= MP) return;
    const int hoff = hh * DHEAD;
    const int l15 = lane & 15;
    const int kb0 = (lane >> 4) * 8;

    Frag qf[2];
    {
        const size_t qrow = ((size_t)(b * MP + m0 + l15)) * HDIM + hoff;
        #pragma unroll
        for (int f = 0; f < 2; f++)
            #pragma unroll
            for (int j = 0; j < 4; j++) {
                qf[f].u[j]     = *(const unsigned int*)&Q[qrow + f * 32 + kb0 + 2 * j];
                qf[f].u[4 + j] = *(const unsigned int*)&Q[qrow + f * 32 + kb0 + 16 + 2 * j];
            }
    }

    Acc o[4];
    for (int f = 0; f < 4; f++) for (int r = 0; r < 8; r++) o[f].f[r] = 0.f;
    float mrow[8], lrow[8];
    for (int r = 0; r < 8; r++) { mrow[r] = -1e30f; lrow[r] = 0.f; }

    const int nkb = (m0 + 16 + 31) / 32;
    for (int kbi = 0; kbi < nkb; kbi++) {
        const int kstart = kbi * 32;

        for (int i = lane; i < 2048; i += 32) {
            const int kk = i >> 6, d = i & 63;
            Vt[w][d][kk] = V[((size_t)(b * MP + kstart + kk)) * HDIM + hoff + d];
        }

        Acc s[2];
        #pragma unroll
        for (int nt = 0; nt < 2; nt++) {
            for (int r = 0; r < 8; r++) s[nt].f[r] = 0.f;
            Frag kf[2];
            const size_t krow =
                ((size_t)(b * MP + kstart + nt * 16 + l15)) * HDIM + hoff;
            #pragma unroll
            for (int f = 0; f < 2; f++)
                #pragma unroll
                for (int j = 0; j < 4; j++) {
                    kf[f].u[j]     = *(const unsigned int*)&Kb[krow + f * 32 + kb0 + 2 * j];
                    kf[f].u[4 + j] = *(const unsigned int*)&Kb[krow + f * 32 + kb0 + 16 + 2 * j];
                }
            s[nt].v = __builtin_amdgcn_wmma_f32_16x16x32_bf16(
                false, qf[0].v, false, kf[0].v, (short)0, s[nt].v, false, false);
            s[nt].v = __builtin_amdgcn_wmma_f32_16x16x32_bf16(
                false, qf[1].v, false, kf[1].v, (short)0, s[nt].v, false, false);
        }

        #pragma unroll
        for (int r = 0; r < 8; r++) {
            const int row = m0 + r + ((lane >> 4) << 3);
            float s0 = s[0].f[r] * 0.125f;
            float s1 = s[1].f[r] * 0.125f;
            if (kstart + l15 > row)      s0 = -1e30f;
            if (kstart + 16 + l15 > row) s1 = -1e30f;
            float mc = fmaxf(s0, s1);
            #pragma unroll
            for (int d = 1; d < 16; d <<= 1) mc = fmaxf(mc, __shfl_xor(mc, d, 32));
            const float mn  = fmaxf(mrow[r], mc);
            const float fac = __expf(mrow[r] - mn);
            const float p0  = __expf(s0 - mn);
            const float p1  = __expf(s1 - mn);
            float ps = p0 + p1;
            #pragma unroll
            for (int d = 1; d < 16; d <<= 1) ps += __shfl_xor(ps, d, 32);
            lrow[r] = lrow[r] * fac + ps;
            mrow[r] = mn;
            #pragma unroll
            for (int f = 0; f < 4; f++) o[f].f[r] *= fac;
            const int rl = r + ((lane >> 4) << 3);
            Pb[w][rl][l15]      = f2bf(p0);
            Pb[w][rl][16 + l15] = f2bf(p1);
        }

        Frag pf;
        #pragma unroll
        for (int j = 0; j < 4; j++) {
            pf.u[j]     = *(const unsigned int*)&Pb[w][l15][kb0 + 2 * j];
            pf.u[4 + j] = *(const unsigned int*)&Pb[w][l15][kb0 + 16 + 2 * j];
        }
        #pragma unroll
        for (int f = 0; f < 4; f++) {
            Frag vf;
            #pragma unroll
            for (int j = 0; j < 4; j++) {
                vf.u[j]     = *(const unsigned int*)&Vt[w][f * 16 + l15][kb0 + 2 * j];
                vf.u[4 + j] = *(const unsigned int*)&Vt[w][f * 16 + l15][kb0 + 16 + 2 * j];
            }
            o[f].v = __builtin_amdgcn_wmma_f32_16x16x32_bf16(
                false, pf.v, false, vf.v, (short)0, o[f].v, false, false);
        }
    }

    #pragma unroll
    for (int f = 0; f < 4; f++)
        #pragma unroll
        for (int r = 0; r < 8; r++) {
            const int row = m0 + r + ((lane >> 4) << 3);
            O[((size_t)(b * MP + row)) * HDIM + hoff + f * 16 + l15] =
                o[f].f[r] / lrow[r];
        }
}

// ---------------------------------------------------------------------------
__global__ __launch_bounds__(256)
void ln_kernel(float* __restrict__ h, const float* __restrict__ sc,
               const float* __restrict__ bi)
{
    __shared__ float rs[256], rq[256];
    const int tid = threadIdx.x;
    float* p = h + (size_t)blockIdx.x * HDIM;
    float s = 0.f, s2 = 0.f;
    for (int i = tid; i < HDIM; i += 256) { const float v = p[i]; s += v; s2 += v * v; }
    rs[tid] = s; rq[tid] = s2; __syncthreads();
    for (int off = 128; off > 0; off >>= 1) {
        if (tid < off) { rs[tid] += rs[tid + off]; rq[tid] += rq[tid + off]; }
        __syncthreads();
    }
    const float mean = rs[0] * (1.f / HDIM);
    const float var  = rq[0] * (1.f / HDIM) - mean * mean;
    const float inv  = rsqrtf(var + LNEPS);
    for (int i = tid; i < HDIM; i += 256)
        p[i] = (p[i] - mean) * inv * sc[i] + bi[i];
}

__global__ __launch_bounds__(256)
void embed_kernel(const float* __restrict__ states, const float* __restrict__ actions,
                  const float* __restrict__ Ws, const float* __restrict__ bs,
                  const float* __restrict__ Wa, const float* __restrict__ ba,
                  const float* __restrict__ pos, float* __restrict__ h)
{
    const int t = blockIdx.x % MAXT;
    const int b = blockIdx.x / MAXT;
    const int col = blockIdx.y * 256 + threadIdx.x;
    float acc;
    if (t == 0) {
        const float* st = states + (size_t)b * TSEQ * SDIM;
        acc = bs[col];
        for (int k = 0; k < SDIM; k++) acc += st[k] * Ws[k * HDIM + col];
    } else {
        const float* ac = actions + ((size_t)b * TSEQ + (t - 1)) * ADIM;
        acc = ba[col];
        for (int k = 0; k < ADIM; k++) acc += ac[k] * Wa[k * HDIM + col];
    }
    acc += pos[(size_t)t * HDIM + col];
    h[((size_t)b * MP + t) * HDIM + col] = acc;
}

__global__ __launch_bounds__(256)
void zero_pad(float* __restrict__ h)
{
    const int per = (MP - MAXT) * HDIM;
    const int idx = blockIdx.x * 256 + threadIdx.x;
    if (idx >= BDIM * per) return;
    const int b = idx / per, rem = idx % per;
    h[((size_t)b * MP + MAXT + rem / HDIM) * HDIM + (rem % HDIM)] = 0.f;
}

__global__ __launch_bounds__(32)
void out_kernel(const float* __restrict__ h, const float* __restrict__ Wout,
                const float* __restrict__ bout, float* __restrict__ out)
{
    const int bt = blockIdx.x;
    const int b = bt / TSEQ, t = bt % TSEQ;
    const int j = threadIdx.x;
    const float* row = h + ((size_t)b * MP + 1 + t) * HDIM;
    float acc = bout[j];
    for (int k = 0; k < HDIM; k++) acc += row[k] * Wout[k * OUTD + j];
    out[((size_t)b * TSEQ + t) * OUTD + j] = acc;
}

// ---------------------------------------------------------------------------
extern "C" void kernel_launch(void* const* d_in, const int* in_sizes, int n_in,
                              void* d_out, int out_size, void* d_ws, size_t ws_size,
                              hipStream_t stream)
{
    const float* states  = (const float*)d_in[1];
    const float* actions = (const float*)d_in[3];
    const float* pos     = (const float*)d_in[8];
    const float* Ws      = (const float*)d_in[9];
    const float* bs      = (const float*)d_in[10];
    const float* Wa      = (const float*)d_in[11];
    const float* ba      = (const float*)d_in[12];
    const float* ln0_s   = (const float*)d_in[13];
    const float* ln0_b   = (const float*)d_in[14];
    const float* Wq = (const float*)d_in[15]; const float* bq = (const float*)d_in[16];
    const float* Wk = (const float*)d_in[17]; const float* bk = (const float*)d_in[18];
    const float* Wv = (const float*)d_in[19]; const float* bv = (const float*)d_in[20];
    const float* Wo = (const float*)d_in[21]; const float* bo = (const float*)d_in[22];
    const float* l1s = (const float*)d_in[23]; const float* l1b = (const float*)d_in[24];
    const float* W1 = (const float*)d_in[25]; const float* b1 = (const float*)d_in[26];
    const float* W2 = (const float*)d_in[27]; const float* b2 = (const float*)d_in[28];
    const float* l2s = (const float*)d_in[29]; const float* l2b = (const float*)d_in[30];
    const float* Wout = (const float*)d_in[31]; const float* bout = (const float*)d_in[32];

    char* ws = (char*)d_ws;
    size_t off = 0;
    float* h = (float*)(ws + off);            off += (size_t)MROWS * HDIM * 4;
    unsigned short* qb = (unsigned short*)(ws + off); off += (size_t)MROWS * HDIM * 2;
    unsigned short* kb = (unsigned short*)(ws + off); off += (size_t)MROWS * HDIM * 2;
    unsigned short* vb = (unsigned short*)(ws + off); off += (size_t)MROWS * HDIM * 2;
    float* attn = (float*)(ws + off);         off += (size_t)MROWS * HDIM * 4;
    unsigned short* m1 = (unsigned short*)(ws + off); off += (size_t)MROWS * FFDIM * 2;
    unsigned short* wt = (unsigned short*)(ws + off);

    const size_t HH = (size_t)HDIM * HDIM;          // 1,048,576
    const size_t HF = (size_t)HDIM * FFDIM;         // 4,194,304
    const size_t LBLK = 4 * HH + 2 * HF;            // bf16 elems per layer
    const size_t need = off + LBLK * NLAYER * 2;    // bytes incl. weight cache
    const bool use_tdm = (ws_size >= need);

    const dim3 blk(256);
    const dim3 gH(HDIM / 256, MROWS / 64);
    const dim3 gF(FFDIM / 256, MROWS / 64);

    if (use_tdm) {                                  // one-time weight prep
        for (int i = 0; i < NLAYER; i++) {
            unsigned short* wl = wt + (size_t)i * LBLK;
            const size_t wOff = (size_t)i * HH;
            wconv_kernel<<<dim3(16, 16), blk, 0, stream>>>(Wq + wOff, wl,          HDIM, HDIM);
            wconv_kernel<<<dim3(16, 16), blk, 0, stream>>>(Wk + wOff, wl + HH,     HDIM, HDIM);
            wconv_kernel<<<dim3(16, 16), blk, 0, stream>>>(Wv + wOff, wl + 2 * HH, HDIM, HDIM);
            wconv_kernel<<<dim3(16, 16), blk, 0, stream>>>(Wo + wOff, wl + 3 * HH, HDIM, HDIM);
            wconv_kernel<<<dim3(64, 16), blk, 0, stream>>>(W1 + (size_t)i * HF, wl + 4 * HH, HDIM, FFDIM);
            wconv_kernel<<<dim3(16, 64), blk, 0, stream>>>(W2 + (size_t)i * HF, wl + 4 * HH + HF, FFDIM, HDIM);
        }
    }

    embed_kernel<<<dim3(BDIM * MAXT, HDIM / 256), blk, 0, stream>>>(
        states, actions, Ws, bs, Wa, ba, pos, h);
    zero_pad<<<(BDIM * (MP - MAXT) * HDIM + 255) / 256, blk, 0, stream>>>(h);
    ln_kernel<<<MROWS, blk, 0, stream>>>(h, ln0_s, ln0_b);

    for (int i = 0; i < NLAYER; i++) {
        const size_t wOff = (size_t)i * HH;
        const size_t bOff = (size_t)i * HDIM;
        unsigned short* wl = wt + (size_t)i * LBLK;

        if (use_tdm) {
            gemm_wmma_tdm<false, true, false, false><<<gH, blk, 0, stream>>>(
                h, wl, bq + bOff, nullptr, qb, MROWS, HDIM, HDIM);
            gemm_wmma_tdm<false, true, false, false><<<gH, blk, 0, stream>>>(
                h, wl + HH, bk + bOff, nullptr, kb, MROWS, HDIM, HDIM);
            gemm_wmma_tdm<false, true, false, false><<<gH, blk, 0, stream>>>(
                h, wl + 2 * HH, bv + bOff, nullptr, vb, MROWS, HDIM, HDIM);
        } else {
            gemm_wmma<false, true, false, false><<<gH, blk, 0, stream>>>(
                h, Wq + wOff, bq + bOff, nullptr, qb, MROWS, HDIM, HDIM);
            gemm_wmma<false, true, false, false><<<gH, blk, 0, stream>>>(
                h, Wk + wOff, bk + bOff, nullptr, kb, MROWS, HDIM, HDIM);
            gemm_wmma<false, true, false, false><<<gH, blk, 0, stream>>>(
                h, Wv + wOff, bv + bOff, nullptr, vb, MROWS, HDIM, HDIM);
        }

        attn_wmma<<<dim3(BDIM * NHEADS, (MP + 127) / 128), blk, 0, stream>>>(
            qb, kb, vb, attn);

        if (use_tdm) {
            gemm_wmma_tdm<false, false, false, true><<<gH, blk, 0, stream>>>(
                attn, wl + 3 * HH, bo + bOff, h, h, MROWS, HDIM, HDIM);
        } else {
            gemm_wmma<false, false, false, true><<<gH, blk, 0, stream>>>(
                attn, Wo + wOff, bo + bOff, h, h, MROWS, HDIM, HDIM);
        }
        ln_kernel<<<MROWS, blk, 0, stream>>>(h, l1s + bOff, l1b + bOff);

        if (use_tdm) {
            gemm_wmma_tdm<false, true, true, false><<<gF, blk, 0, stream>>>(
                h, wl + 4 * HH, b1 + (size_t)i * FFDIM, nullptr, m1,
                MROWS, FFDIM, HDIM);
            gemm_wmma_tdm<true, false, false, true><<<gH, blk, 0, stream>>>(
                m1, wl + 4 * HH + HF, b2 + bOff, h, h, MROWS, HDIM, FFDIM);
        } else {
            gemm_wmma<false, true, true, false><<<gF, blk, 0, stream>>>(
                h, W1 + (size_t)i * HF, b1 + (size_t)i * FFDIM, nullptr, m1,
                MROWS, FFDIM, HDIM);
            gemm_wmma<true, false, false, true><<<gH, blk, 0, stream>>>(
                m1, W2 + (size_t)i * HF, b2 + bOff, h, h, MROWS, HDIM, FFDIM);
        }
        ln_kernel<<<MROWS, blk, 0, stream>>>(h, l2s + bOff, l2b + bOff);
    }

    out_kernel<<<BDIM * TSEQ, dim3(32), 0, stream>>>(h, Wout, bout, (float*)d_out);
}
